// GVPModel_64931315580984
// MI455X (gfx1250) — compile-verified, matrix operand order checked by reading, not run on hardware
//
#include <hip/hip_runtime.h>

typedef __attribute__((ext_vector_type(16))) _Float16 v16h;
typedef __attribute__((ext_vector_type(8)))  float    v8f;

#define LDIM 128
#define TDIM 512
#define H2   256
#define NSC  128
#define MAX_VDW_C 0.0356f
#define MIN_VDW_C 0.0178f
#define DEV_VDW_R 0.2f

#define WAVES_PER_BLOCK 16
#define THREADS_PER_BLOCK (WAVES_PER_BLOCK * 32)
#define W_BYTES 131072          // 65536 f16 weight fragments
#define W_UINT4 (W_BYTES / 16)  // 8192 uint4

// ---------------------------------------------------------------------------
// Prep: convert Wa1|Wb1 (fp32 [256,128] each) into f16 fragment-ordered layout
// matching the CDNA5 16-bit B-matrix (32x16) VGPR layout, and zero accumulator.
// frag f = kt*16+nt ; element e = f*512 + lane*16 + slot (one v16h per lane)
// ---------------------------------------------------------------------------
__global__ __launch_bounds__(256) void decoder_prep(
    const float* __restrict__ Wa1, const float* __restrict__ Wb1,
    _Float16* __restrict__ wf, float* __restrict__ acc)
{
    int e = blockIdx.x * 256 + threadIdx.x;      // 0..65535
    int s  = e & 15;
    int l  = (e >> 4) & 31;
    int f  = e >> 9;
    int kt = f >> 4;
    int nt = f & 15;
    int kbase  = (l >= 16) ? 8 : 0;
    int klocal = kbase + ((s < 8) ? s : (s + 8));
    int c = kt * 32 + klocal;                    // input channel 0..255
    int n = nt * 16 + (l & 15);                  // 0-127 A-net, 128-255 B-net
    float v = (n < NSC) ? Wa1[c * NSC + n] : Wb1[c * NSC + (n - NSC)];
    wf[e] = (_Float16)v;
    if (e < 32) acc[e] = 0.0f;
}

// ---------------------------------------------------------------------------
// Main fused kernel. 512 threads = 16 waves; block stages 128KB of f16 weight
// fragments into LDS once; each wave runs a 16-row M-tile with 128x
// v_wmma_f32_16x16x32_f16 (B triple-buffered from LDS), folds the layer-2
// dot, then reduces geometry energies to 4 atomic adds per wave.
// ---------------------------------------------------------------------------
__global__ __launch_bounds__(THREADS_PER_BLOCK) void decoder_main(
    const float* __restrict__ indice,   // [B,3,L,T]
    const float* __restrict__ lpos,     // [B,L,3]
    const float* __restrict__ tpos,     // [B,T,3]
    const float* __restrict__ lrad,     // [B,L]
    const float* __restrict__ trad,     // [B,T]
    const float* __restrict__ lnm,      // [B,L]
    const float* __restrict__ tnm,      // [B,T]
    const float* __restrict__ hcat,     // [B,L,T,256]
    const float* __restrict__ ba1, const float* __restrict__ Wa2, const float* __restrict__ ba2,
    const float* __restrict__ bb1, const float* __restrict__ Wb2, const float* __restrict__ bb2,
    const float* __restrict__ hbond_c, const float* __restrict__ hydro_c,
    const uint4* __restrict__ wfrag,    // f16 weight fragments in d_ws (8192 uint4)
    float* __restrict__ acc)            // [B*4]
{
    extern __shared__ uint4 smem4[];                 // 8192 uint4 weights + xch
    float* xch = (float*)(smem4 + W_UINT4);          // [16 waves][16 rows][2]

    const int tid  = threadIdx.x;
    const int lane = tid & 31;
    const int wv   = tid >> 5;
    const int grp  = lane >> 4;
    const int ln   = lane & 15;
    const int pairBase = blockIdx.x * (WAVES_PER_BLOCK * 16) + wv * 16;

    // ---- stage weight fragments into LDS (16B vectors, whole block) --------
#pragma unroll
    for (int i = 0; i < W_UINT4 / THREADS_PER_BLOCK; ++i)
        smem4[tid + i * THREADS_PER_BLOCK] = wfrag[tid + i * THREADS_PER_BLOCK];

    // ---- load A fragments: 16 rows x 256 cols of h_cat, f32 -> f16 ---------
    // 16-bit A-matrix 16x32 layout: lane<16 row=lane K={0..7,16..23};
    // lanes 16..31 row=lane-16 K={8..15,24..31} per K-tile of 32.
    const float* rowp = hcat + (size_t)(pairBase + ln) * H2;
    v16h afrag[8];
#pragma unroll
    for (int kt = 0; kt < 8; ++kt) {
        const int off = kt * 32 + grp * 8;
        const float4* r4 = (const float4*)(rowp + off);
        float4 q0 = r4[0];
        float4 q1 = r4[1];
        float4 q2 = r4[4];   // rowp + off + 16
        float4 q3 = r4[5];
        v16h a;
        a[0]  = (_Float16)q0.x; a[1]  = (_Float16)q0.y; a[2]  = (_Float16)q0.z; a[3]  = (_Float16)q0.w;
        a[4]  = (_Float16)q1.x; a[5]  = (_Float16)q1.y; a[6]  = (_Float16)q1.z; a[7]  = (_Float16)q1.w;
        a[8]  = (_Float16)q2.x; a[9]  = (_Float16)q2.y; a[10] = (_Float16)q2.z; a[11] = (_Float16)q2.w;
        a[12] = (_Float16)q3.x; a[13] = (_Float16)q3.y; a[14] = (_Float16)q3.z; a[15] = (_Float16)q3.w;
        afrag[kt] = a;
    }

    __syncthreads();   // weights resident in LDS

    // Two LDS base pointers so every ds_load offset fits the 16-bit immediate:
    // fragment (kt,nt) lives at byte (kt*16+nt)*1024 + lane*32; kt>=4 crosses
    // 64KB, so use a second base at +65536.
    const uint4* sm_lo = smem4 + lane * 2;            // + rel (uint4 units)
    const uint4* sm_hi = smem4 + 4096 + lane * 2;     // +65536 bytes

    union BU { uint4 q2[2]; v16h v; };

    // ---- layer-1 GEMM (N=256 fused, B triple-buffered from LDS) ------------
    float dA[8] = {0,0,0,0,0,0,0,0};
    float dB[8] = {0,0,0,0,0,0,0,0};
#pragma unroll
    for (int nt = 0; nt < 16; ++nt) {
        v8f c = {};
        BU bufs[3];

        // prefetch fragments kt=0,1
#pragma unroll
        for (int p = 0; p < 2; ++p) {
            const uint4* base = (p < 4) ? sm_lo : sm_hi;
            const int rel = ((p & 3) * 16 + nt) * 64;
            bufs[p].q2[0] = base[rel];
            bufs[p].q2[1] = base[rel + 1];
        }

#pragma unroll
        for (int kt = 0; kt < 8; ++kt) {
            if (kt + 2 < 8) {
                const int kp = kt + 2;
                const uint4* base = (kp < 4) ? sm_lo : sm_hi;
                const int rel = ((kp & 3) * 16 + nt) * 64;
                bufs[kp % 3].q2[0] = base[rel];
                bufs[kp % 3].q2[1] = base[rel + 1];
            }
            c = __builtin_amdgcn_wmma_f32_16x16x32_f16(
                    false, afrag[kt], false, bufs[kt % 3].v, (short)0, c, false, false);
        }

        // C layout: lane holds column N = nt*16 + ln; VGPR j holds row j + 8*grp
        const int n = nt * 16 + ln;
        const float b1 = (n < NSC) ? ba1[n] : bb1[n - NSC];
        const float w2 = (n < NSC) ? Wa2[n] : Wb2[n - NSC];
#pragma unroll
        for (int j = 0; j < 8; ++j) {
            float h = c[j] + b1;
            h = (h > 0.0f) ? h : 0.0f;            // ReLU
            if (nt < 8) dA[j] += h * w2;          // A-net columns
            else        dB[j] += h * w2;          // B-net columns
        }
    }

    // ---- reduce layer-2 dots across the 16 lanes of each half-wave ---------
#pragma unroll
    for (int j = 0; j < 8; ++j) {
#pragma unroll
        for (int m = 1; m < 16; m <<= 1) {
            dA[j] += __shfl_xor(dA[j], m, 32);
            dB[j] += __shfl_xor(dB[j], m, 32);
        }
    }

    // lane ln==0 of each group finalizes 8 rows (rows grp*8 .. grp*8+7)
    if (ln == 0) {
        const float ba2v = ba2[0], bb2v = bb2[0];
#pragma unroll
        for (int j = 0; j < 8; ++j) {
            float Av = 1.0f / (1.0f + __expf(-(dA[j] + ba2v)));
            Av = Av * (MAX_VDW_C - MIN_VDW_C) + MIN_VDW_C;
            float Bv = tanhf(dB[j] + bb2v) * DEV_VDW_R;
            xch[(wv * 16 + grp * 8 + j) * 2 + 0] = Av;
            xch[(wv * 16 + grp * 8 + j) * 2 + 1] = Bv;
        }
    }
    __syncthreads();

    // ---- per-pair geometry + energies (lanes 0..15 = rows 0..15) -----------
    float e0 = 0.0f, e1 = 0.0f, e2 = 0.0f, e3 = 0.0f;
    if (lane < 16) {
        const int p  = pairBase + lane;
        const int b  = p >> 16;                 // L*T = 65536
        const int li = (p >> 9) & (LDIM - 1);
        const int ti = p & (TDIM - 1);
        const float Av = xch[(wv * 16 + lane) * 2 + 0];
        const float Bv = xch[(wv * 16 + lane) * 2 + 1];

        const float* lp = lpos + ((size_t)b * LDIM + li) * 3;
        const float* tp = tpos + ((size_t)b * TDIM + ti) * 3;
        const float dx = lp[0] - tp[0];
        const float dy = lp[1] - tp[1];
        const float dz = lp[2] - tp[2];
        float dm = sqrtf(dx * dx + dy * dy + dz * dz + 1e-10f);
        if (dm < 0.5f) dm = 1e10f;

        const float dm0 = lrad[b * LDIM + li] + trad[b * TDIM + ti] + Bv;
        const float d   = dm - dm0;
        const float dm0s = (dm0 < 1e-4f) ? 1.0f : dm0;
        const float ratio = dm0s / dm;
        const float r2 = ratio * ratio;
        const float r6 = r2 * r2 * r2;
        const float ve = fminf(r6 * r6 - 2.0f * r6, 100.0f)
                         * lnm[b * LDIM + li] * tnm[b * TDIM + ti];
        e0 = Av * ve;

        const size_t ib = (size_t)b * 3 * (LDIM * TDIM) + (size_t)li * TDIM + ti;
        const float i0 = indice[ib];
        const float i1 = indice[ib + (size_t)(LDIM * TDIM)];
        const float i2 = indice[ib + (size_t)(2 * LDIM * TDIM)];
        const float hc = hbond_c[0];
        const float yc = hydro_c[0];
        float r;
        r = d * i0 * (-1.0f / 0.7f); r = fminf(fmaxf(r, 0.0f), 1.0f); e1 = r * -(hc * hc);
        r = d * i1 * (-1.0f / 0.7f); r = fminf(fmaxf(r, 0.0f), 1.0f); e2 = r * -(hc * hc);
        r = (-d + 1.5f) * i2;        r = fminf(fmaxf(r, 0.0f), 1.0f); e3 = r * -(yc * yc);
    }

    // reduce 16 pairs of this wave (lanes >=16 hold zeros)
#pragma unroll
    for (int m = 1; m < 16; m <<= 1) {
        e0 += __shfl_xor(e0, m, 32);
        e1 += __shfl_xor(e1, m, 32);
        e2 += __shfl_xor(e2, m, 32);
        e3 += __shfl_xor(e3, m, 32);
    }
    if (lane == 0) {
        const int b = pairBase >> 16;
        atomicAdd(&acc[b * 4 + 0], e0);
        atomicAdd(&acc[b * 4 + 1], e1);
        atomicAdd(&acc[b * 4 + 2], e2);
        atomicAdd(&acc[b * 4 + 3], e3);
    }
}

// ---------------------------------------------------------------------------
// Finalize: rotor penalty, write [B,4]
// ---------------------------------------------------------------------------
__global__ void decoder_finalize(const float* __restrict__ acc,
                                 const float* __restrict__ rotor,
                                 const float* __restrict__ rotor_c,
                                 float* __restrict__ out, int n)
{
    int j = blockIdx.x * blockDim.x + threadIdx.x;
    if (j < n) {
        int b = j >> 2;
        float rc = rotor_c[0];
        out[j] = acc[j] / (1.0f + rc * rc * rotor[b]);
    }
}

extern "C" void kernel_launch(void* const* d_in, const int* in_sizes, int n_in,
                              void* d_out, int out_size, void* d_ws, size_t ws_size,
                              hipStream_t stream)
{
    const float* indice = (const float*)d_in[0];
    const float* lpos   = (const float*)d_in[1];
    const float* tpos   = (const float*)d_in[2];
    const float* rotor  = (const float*)d_in[3];
    const float* lrad   = (const float*)d_in[4];
    const float* trad   = (const float*)d_in[5];
    const float* lnm    = (const float*)d_in[6];
    const float* tnm    = (const float*)d_in[7];
    const float* hcat   = (const float*)d_in[8];
    const float* Wa1    = (const float*)d_in[9];
    const float* ba1    = (const float*)d_in[10];
    const float* Wa2    = (const float*)d_in[11];
    const float* ba2    = (const float*)d_in[12];
    const float* Wb1    = (const float*)d_in[13];
    const float* bb1    = (const float*)d_in[14];
    const float* Wb2    = (const float*)d_in[15];
    const float* bb2    = (const float*)d_in[16];
    const float* hbc    = (const float*)d_in[17];
    const float* hyc    = (const float*)d_in[18];
    const float* roc    = (const float*)d_in[19];

    const int B = in_sizes[3];                       // rotor is [B]
    _Float16* wf  = (_Float16*)d_ws;                 // 65536 f16 = 128 KB
    float*    acc = (float*)((char*)d_ws + W_BYTES);

    // 1) weight fragments (f16) + zero accumulator
    decoder_prep<<<256, 256, 0, stream>>>(Wa1, Wb1, wf, acc);

    // 2) fused decoder: B*L*T/256 blocks, 16 waves each, 16 pairs per wave
    //    dynamic LDS: 128KB weights + 2KB row exchange
    const int nblocks = B * (LDIM * TDIM) / (WAVES_PER_BLOCK * 16);   // B*256
    const size_t shmem = W_BYTES + WAVES_PER_BLOCK * 16 * 2 * sizeof(float);
    decoder_main<<<nblocks, THREADS_PER_BLOCK, shmem, stream>>>(
        indice, lpos, tpos, lrad, trad, lnm, tnm, hcat,
        ba1, Wa2, ba2, bb1, Wb2, bb2, hbc, hyc,
        (const uint4*)wf, acc);

    // 3) rotor penalty -> d_out [B,4]
    decoder_finalize<<<1, 64, 0, stream>>>(acc, rotor, roc, (float*)d_out, B * 4);
}
